// NormalConvolution2D_46488726012099
// MI455X (gfx1250) — matrix operand order repaired and checked by reading, not compiled, not run on hardware
//
#include <hip/hip_runtime.h>

typedef __attribute__((ext_vector_type(16))) _Float16 v16h;
typedef __attribute__((ext_vector_type(8)))  _Float16 v8h;
typedef __attribute__((ext_vector_type(8)))  float    v8f;

constexpr int C_IN   = 16;
constexpr int F_OUT  = 32;
constexpr int HWDIM  = 256;
constexpr int ROWS   = 4;              // output rows per block
constexpr int SROWS  = ROWS + 2;       // staged input rows (halo)
constexpr int COLS   = HWDIM + 2;      // padded columns
constexpr int KPAD   = 160;            // 144 -> 5 chunks of K=32
constexpr int NCHUNK = 5;

__device__ __forceinline__ float sigm(float v) {
    return 1.0f / (1.0f + __expf(-v));
}

__global__ __launch_bounds__(256)
void conv3x3_wmma_f16(const float* __restrict__ x,
                      const float* __restrict__ wgt,
                      const float* __restrict__ bias,
                      float* __restrict__ out)
{
    // input rows h0-1 .. h0+ROWS as f16, channel-last: [r][colp][c]
    __shared__ _Float16 sIn[SROWS * COLS * C_IN];      // 24768 halfs = 49536 B
    // weights, K permuted to (tap, c) order, zero padded: [n][k]
    __shared__ _Float16 sB[F_OUT * KPAD];              // 5120 halfs = 10240 B

    const int tid  = threadIdx.x;
    const int lane = tid & 31;             // wave32
    const int wave = tid >> 5;
    const int h0   = blockIdx.x * ROWS;    // 64 row-groups
    const int b    = blockIdx.y;           // 16 batches

    // ---- stage input (fp32 -> f16, zero-padded halo), coalesced on colp ----
    for (int idx = tid; idx < SROWS * C_IN * COLS; idx += 256) {
        int colp = idx % COLS;
        int t    = idx / COLS;
        int c    = t % C_IN;
        int r    = t / C_IN;
        int col  = colp - 1;
        int grow = h0 - 1 + r;
        float v = 0.0f;
        if ((unsigned)col < (unsigned)HWDIM && (unsigned)grow < (unsigned)HWDIM)
            v = x[(((b * C_IN) + c) * HWDIM + grow) * HWDIM + col];
        sIn[(r * COLS + colp) * C_IN + c] = (_Float16)v;
    }

    // ---- stage weights: K_new = tap*16 + c  (tap = dh*3+dw), zero pad ----
    for (int idx = tid; idx < F_OUT * KPAD; idx += 256) {
        int k = idx % KPAD;
        int n = idx / KPAD;
        float v = 0.0f;
        if (k < C_IN * 9) {
            int tap = k >> 4;
            int c   = k & 15;
            v = wgt[(c * 9 + tap) * F_OUT + n];   // original rows are (c, kh, kw)
        }
        sB[n * KPAD + k] = (_Float16)v;
    }

    __syncthreads();

    // ---- preload all B fragments into registers (ISA 32x16 16-bit B layout:
    //      lanes 0-15 hold K 0..7,16..23, lanes 16-31 hold K 8..15,24..31, N = lane&15) ----
    v16h bf[NCHUNK][2];
    {
        const int n_lo = lane & 15;
        const int kofs = (lane >> 4) ? 8 : 0;
#pragma unroll
        for (int j = 0; j < NCHUNK; ++j) {
#pragma unroll
            for (int nh = 0; nh < 2; ++nh) {
                const _Float16* p = &sB[(nh * 16 + n_lo) * KPAD + j * 32 + kofs];
                v8h lo = *(const v8h*)(p);
                v8h hi = *(const v8h*)(p + 16);
                bf[j][nh] = __builtin_shufflevector(
                    lo, hi, 0,1,2,3,4,5,6,7,8,9,10,11,12,13,14,15);
            }
        }
    }

    const int m  = lane & 15;              // A row = output pixel in strip
    const int co = (lane >> 4) << 3;       // channel offset 0 / 8 (A layout halves)

    // ---- 64 strips (ROWS rows x 16 w-strips of 16 px) over 8 waves ----
    for (int s = wave; s < ROWS * 16; s += 8) {
        const int rr = s >> 4;
        const int w0 = (s & 15) << 4;
        const int h  = h0 + rr;

        // --- software pipeline: issue ALL A-fragment loads first (9x
        //     independent ds_load_b128), single dscnt wait, then 10 WMMAs ---
        v16h afrag[NCHUNK];
#pragma unroll
        for (int j = 0; j < NCHUNK - 1; ++j) {
            const int t0 = 2 * j;
            const int t1 = 2 * j + 1;
            // A fragment per ISA 16-bit 16x32 layout: lane L -> row M = L&15,
            // halves 0..7 = tap t0 ch co..co+7, halves 8..15 = tap t1 ch co..co+7
            const _Float16* p0 =
                &sIn[((rr + t0 / 3) * COLS + (w0 + m + t0 % 3)) * C_IN + co];
            const _Float16* p1 =
                &sIn[((rr + t1 / 3) * COLS + (w0 + m + t1 % 3)) * C_IN + co];
            v8h lo = *(const v8h*)p0;                 // ds_load_b128
            v8h hi = *(const v8h*)p1;                 // ds_load_b128
            afrag[j] = __builtin_shufflevector(
                lo, hi, 0,1,2,3,4,5,6,7,8,9,10,11,12,13,14,15);
        }
        {
            // padded chunk: K 128..143 = tap 8, K 144..159 multiplies a zero B
            const _Float16* p0 =
                &sIn[((rr + 2) * COLS + (w0 + m + 2)) * C_IN + co];  // tap 8
            v8h lo = *(const v8h*)p0;                 // single ds_load_b128
            afrag[NCHUNK - 1] = __builtin_shufflevector(
                lo, lo, 0,1,2,3,4,5,6,7,0,1,2,3,4,5,6,7);
        }

        v8f acc0 = {};
        v8f acc1 = {};
#pragma unroll
        for (int j = 0; j < NCHUNK; ++j) {
            acc0 = __builtin_amdgcn_wmma_f32_16x16x32_f16(
                false, afrag[j], false, bf[j][0], (short)0, acc0, false, false);
            acc1 = __builtin_amdgcn_wmma_f32_16x16x32_f16(
                false, afrag[j], false, bf[j][1], (short)0, acc1, false, false);
        }

        // ---- epilogue: C/D layout -> lane owns 8 consecutive w per filter ----
        const int wb = w0 + ((lane >> 4) << 3);       // w base 0 / +8
#pragma unroll
        for (int nh = 0; nh < 2; ++nh) {
            const int f = nh * 16 + (lane & 15);
            const v8f acc = nh ? acc1 : acc0;
            const float* bp = bias + (f * HWDIM + h) * HWDIM + wb;
            float*       op = out  + (((b * F_OUT + f) * HWDIM) + h) * HWDIM + wb;
            float4 bb0 = *(const float4*)bp;
            float4 bb1 = *(const float4*)(bp + 4);
            float4 o0, o1;
            o0.x = sigm(acc[0] + bb0.x);
            o0.y = sigm(acc[1] + bb0.y);
            o0.z = sigm(acc[2] + bb0.z);
            o0.w = sigm(acc[3] + bb0.w);
            o1.x = sigm(acc[4] + bb1.x);
            o1.y = sigm(acc[5] + bb1.y);
            o1.z = sigm(acc[6] + bb1.z);
            o1.w = sigm(acc[7] + bb1.w);
            *(float4*)op       = o0;                  // global_store_b128
            *(float4*)(op + 4) = o1;                  // global_store_b128
        }
    }
}

extern "C" void kernel_launch(void* const* d_in, const int* in_sizes, int n_in,
                              void* d_out, int out_size, void* d_ws, size_t ws_size,
                              hipStream_t stream) {
    const float* x    = (const float*)d_in[0];   // (16, 16, 256, 256) f32
    const float* wgt  = (const float*)d_in[1];   // (144, 32) f32
    const float* bias = (const float*)d_in[2];   // (32, 256, 256) f32
    float* out = (float*)d_out;                  // (16, 32, 256, 256) f32

    dim3 grid(HWDIM / ROWS, 16);                 // 64 row-groups x 16 batches
    conv3x3_wmma_f16<<<grid, 256, 0, stream>>>(x, wgt, bias, out);
}
